// MLPResNetBlock_Pro_68719477557
// MI455X (gfx1250) — matrix operand order repaired and checked by reading, not compiled, not run on hardware
//
#include <hip/hip_runtime.h>
#include <hip/hip_bf16.h>
#include <math.h>

// ---------------------------------------------------------------------------
// MI455X (gfx1250) implementation.
// GEMMs (123 GFLOP total, dominated by h_t projections) run on the bf16 WMMA
// path (v_wmma_f32_16x16x32_bf16, f32 accumulate). Big intermediates kept in
// bf16 to halve HBM traffic (23.3 TB/s roofline ~20us). K-loop uses the CDNA5
// async global->LDS DMA (ASYNCcnt) with double-buffered LDS tiles: one
// barrier per K-step, tile k+1 in flight while WMMAs consume tile k.
// Attention/RoPE/LN are <1% of FLOPs and run as plain wave32 VALU kernels.
// ---------------------------------------------------------------------------

typedef __bf16 bf16t;
typedef __attribute__((ext_vector_type(16))) __bf16 v16bf;
typedef __attribute__((ext_vector_type(8)))  float  v8f;
typedef __attribute__((ext_vector_type(4)))  int    v4i;

static constexpr int kDIM  = 896;
static constexpr int kNH   = 8;
static constexpr int kHD   = 112;
static constexpr int kB    = 64;
static constexpr int kT    = 8;
static constexpr int kKA   = 65;   // 64 + p
static constexpr int kKT   = 512;
static constexpr int kNKEY = kT + kKA + kKT;  // 585

#if __has_builtin(__builtin_amdgcn_global_load_async_to_lds_b128)
#define HAVE_ASYNC_LDS 1
#else
#define HAVE_ASYNC_LDS 0
#endif

// Builtin signature (from hipcc diagnostic): param 0 is v4i* in addrspace(1)
// (printed as "int4 __device__*"); LDS side is the addrspace(3) counterpart.
#define ASGLOBAL(p) ((__attribute__((address_space(1))) v4i*)(p))
#define ASLDS(p)    ((__attribute__((address_space(3))) v4i*)(p))

__device__ __forceinline__ void wait_async_zero() {
#if __has_builtin(__builtin_amdgcn_s_wait_asynccnt)
  __builtin_amdgcn_s_wait_asynccnt(0);
#else
  asm volatile("s_wait_asynccnt 0x0" ::: "memory");
#endif
}

// ---------------- weight transpose + f32->bf16 ------------------------------
// Produces W^T (N-major) in bf16 so GEMM B-tiles are contiguous in K.
__global__ void __launch_bounds__(256) wtrans_kernel(const float* __restrict__ W,
                                                     bf16t* __restrict__ Wt) {
  __shared__ float tile[32][33];
  const int tx = threadIdx.x & 31, ty = threadIdx.x >> 5;  // 32x8
  const int k0 = blockIdx.x * 32, n0 = blockIdx.y * 32;
#pragma unroll
  for (int i = 0; i < 32; i += 8)
    tile[ty + i][tx] = W[(size_t)(k0 + ty + i) * kDIM + n0 + tx];
  __syncthreads();
#pragma unroll
  for (int i = 0; i < 32; i += 8)
    Wt[(size_t)(n0 + ty + i) * kDIM + k0 + tx] = (bf16t)tile[tx][ty + i];
}

// ---------------- elementwise f32 -> bf16 -----------------------------------
__global__ void f2bf_kernel(const float* __restrict__ in, bf16t* __restrict__ out,
                            long n) {
  long i = (long)blockIdx.x * blockDim.x + threadIdx.x;
  const long stride = (long)gridDim.x * blockDim.x;
  for (; i < n; i += stride) out[i] = (bf16t)in[i];
}

// ---------------- build h_ad = concat(h_a, p) in bf16 -----------------------
__global__ void had_build_kernel(const float* __restrict__ h_a,
                                 const float* __restrict__ p,
                                 bf16t* __restrict__ out) {
  const long total = (long)kB * kKA * kDIM;
  long i = (long)blockIdx.x * blockDim.x + threadIdx.x;
  const long stride = (long)gridDim.x * blockDim.x;
  for (; i < total; i += stride) {
    long row = i / kDIM;
    int  c   = (int)(i % kDIM);
    int  b   = (int)(row / kKA);
    int  j   = (int)(row % kKA);
    float v = (j < kKA - 1) ? h_a[((size_t)b * (kKA - 1) + j) * kDIM + c]
                            : p[(size_t)b * kDIM + c];
    out[i] = (bf16t)v;
  }
}

// ---------------- WMMA GEMM: C[M,896] = A[M,896] @ W + bias -----------------
// A: bf16 row-major. Bt: bf16 W^T (row n holds column n of W, contiguous K).
// Workgroup tile 128x128, 8 waves (2 in M x 4 in N), wave tile 64x32,
// K-step 32 per v_wmma_f32_16x16x32_bf16. Fragment layouts per CDNA5 ISA 7.12.2.
// Async path: GLOBAL_LOAD_ASYNC_TO_LDS_B128 + double buffer, 1 barrier/K-step.
template <int RELU, typename OutT>
__global__ void __launch_bounds__(256) gemm_wmma_bf16(
    const bf16t* __restrict__ A, const bf16t* __restrict__ Bt,
    const float* __restrict__ bias, OutT* __restrict__ C, int M) {
  __shared__ bf16t As[2][128][40];  // pad 32->40 to spread banks
  __shared__ bf16t Bs[2][128][40];

  const int tid  = threadIdx.x;
  const int lane = tid & 31;
  const int wave = tid >> 5;
  const int mBlock = blockIdx.x * 128;
  const int nBlock = blockIdx.y * 128;
  const int waveM = (wave & 1) * 64;
  const int waveN = (wave >> 1) * 32;

  v8f acc[4][2];
#pragma unroll
  for (int i = 0; i < 4; ++i)
#pragma unroll
    for (int j = 0; j < 2; ++j) acc[i][j] = v8f{0.f, 0.f, 0.f, 0.f, 0.f, 0.f, 0.f, 0.f};

  const int ldrow = tid >> 1;          // 0..127
  const int ldcol = (tid & 1) * 16;    // 0 or 16 (bf16 elements)
  const bool aValid = (mBlock + ldrow) < M;
  const size_t aBase = (size_t)(mBlock + ldrow) * kDIM;
  const size_t bBase = (size_t)(nBlock + ldrow) * kDIM;

  union Frag { uint4 u[2]; v16bf v; };

  // Per-K-step compute on buffer `bb`.
  auto compute_tile = [&](int bb) {
    // B fragments: lane<16 -> column n, K{0..7,16..23}; lane>=16 -> K{8..15,24..31}
    Frag bfr[2];
#pragma unroll
    for (int ni = 0; ni < 2; ++ni) {
      const int brow = waveN + ni * 16 + (lane & 15);
      const int koff = (lane >> 4) * 8;
      bfr[ni].u[0] = *reinterpret_cast<const uint4*>(&Bs[bb][brow][koff]);
      bfr[ni].u[1] = *reinterpret_cast<const uint4*>(&Bs[bb][brow][koff + 16]);
    }
    // A fragments: lane<16 -> row m, K{0..7,16..23}; lane>=16 -> K{8..15,24..31}
#pragma unroll
    for (int mi = 0; mi < 4; ++mi) {
      Frag afr;
      const int arow = waveM + mi * 16 + (lane & 15);
      const int ksel = (lane >> 4) * 8;
      afr.u[0] = *reinterpret_cast<const uint4*>(&As[bb][arow][ksel]);
      afr.u[1] = *reinterpret_cast<const uint4*>(&As[bb][arow][ksel + 16]);
#pragma unroll
      for (int ni = 0; ni < 2; ++ni)
        acc[mi][ni] = __builtin_amdgcn_wmma_f32_16x16x32_bf16(
            false, afr.v, false, bfr[ni].v, (short)0, acc[mi][ni], false, false);
    }
  };

#if HAVE_ASYNC_LDS
  // Async DMA global->LDS, double buffered. Rows >= M leave stale LDS; those
  // only affect output rows the epilogue already guards.
  auto issue_tile = [&](int kk, int bb) {
    if (aValid) {
      __builtin_amdgcn_global_load_async_to_lds_b128(
          ASGLOBAL(A + aBase + kk + ldcol), ASLDS(&As[bb][ldrow][ldcol]), 0, 0);
      __builtin_amdgcn_global_load_async_to_lds_b128(
          ASGLOBAL(A + aBase + kk + ldcol + 8), ASLDS(&As[bb][ldrow][ldcol + 8]), 0, 0);
    }
    __builtin_amdgcn_global_load_async_to_lds_b128(
        ASGLOBAL(Bt + bBase + kk + ldcol), ASLDS(&Bs[bb][ldrow][ldcol]), 0, 0);
    __builtin_amdgcn_global_load_async_to_lds_b128(
        ASGLOBAL(Bt + bBase + kk + ldcol + 8), ASLDS(&Bs[bb][ldrow][ldcol + 8]), 0, 0);
  };

  issue_tile(0, 0);
  wait_async_zero();
  __syncthreads();
  int buf = 0;
  for (int k0 = 0; k0 < kDIM; k0 += 32) {
    if (k0 + 32 < kDIM) issue_tile(k0 + 32, buf ^ 1);  // prefetch next tile
    compute_tile(buf);
    wait_async_zero();   // own async ops for next tile done
    __syncthreads();     // all waves: next tile ready, current tile consumed
    buf ^= 1;
  }
#else
  // Fallback: stage through VGPRs, single buffer, two barriers per K-step.
  for (int k0 = 0; k0 < kDIM; k0 += 32) {
    uint4 a0 = make_uint4(0, 0, 0, 0), a1 = make_uint4(0, 0, 0, 0);
    if (aValid) {
      const uint4* s = reinterpret_cast<const uint4*>(A + aBase + k0 + ldcol);
      a0 = s[0]; a1 = s[1];
    }
    const uint4* bs = reinterpret_cast<const uint4*>(Bt + bBase + k0 + ldcol);
    uint4 b0 = bs[0], b1 = bs[1];

    __syncthreads();
    *reinterpret_cast<uint4*>(&As[0][ldrow][ldcol])     = a0;
    *reinterpret_cast<uint4*>(&As[0][ldrow][ldcol + 8]) = a1;
    *reinterpret_cast<uint4*>(&Bs[0][ldrow][ldcol])     = b0;
    *reinterpret_cast<uint4*>(&Bs[0][ldrow][ldcol + 8]) = b1;
    __syncthreads();
    compute_tile(0);
  }
#endif

  // Epilogue. C/D layout: VGPR r -> M = r + (lane>=16 ? 8 : 0), N = lane&15.
#pragma unroll
  for (int mi = 0; mi < 4; ++mi) {
#pragma unroll
    for (int ni = 0; ni < 2; ++ni) {
      const int col = nBlock + waveN + ni * 16 + (lane & 15);
      const float bc = bias[col];
#pragma unroll
      for (int r = 0; r < 8; ++r) {
        const int row = mBlock + waveM + mi * 16 + ((lane >> 4) << 3) + r;
        if (row < M) {
          float v = acc[mi][ni][r] + bc;
          if (RELU) v = fmaxf(v, 0.f);
          C[(size_t)row * kDIM + col] = (OutT)v;
        }
      }
    }
  }
}

// ---------------- RoPE (reference's mixed convention), in-place bf16 --------
// theta(d) = pos * 10000^(-2*(d%56)/112); out[2i]=x0*cos(t0)-x1*sin(t0),
// out[2i+1]=x1*cos(t1)+x0*sin(t1) with t0,t1 at emb indices d0,d1.
__global__ void rope_kernel(bf16t* __restrict__ t, int M, int L) {
  const long total = (long)M * (kDIM / 2);
  long i = (long)blockIdx.x * blockDim.x + threadIdx.x;
  if (i >= total) return;
  const int row  = (int)(i / (kDIM / 2));
  const int pr   = (int)(i % (kDIM / 2));
  const int head = pr / (kHD / 2);
  const int dp   = pr % (kHD / 2);
  const int pos  = row % L;
  const int d0 = 2 * dp, d1 = d0 + 1;
  const float c = 0.16447036f;  // ln(10000)/56
  const float th0 = (float)pos * expf(-(float)(d0 % 56) * c);
  const float th1 = (float)pos * expf(-(float)(d1 % 56) * c);
  const size_t base = (size_t)row * kDIM + head * kHD + d0;
  const float x0 = (float)t[base], x1 = (float)t[base + 1];
  t[base]     = (bf16t)(x0 * cosf(th0) - x1 * sinf(th0));
  t[base + 1] = (bf16t)(x1 * cosf(th1) + x0 * sinf(th1));
}

// ---------------- attention: one wave per (b,h,t), 585 keys -----------------
__global__ void __launch_bounds__(256) attn_kernel(
    const bf16t* __restrict__ q,  const bf16t* __restrict__ ks,
    const bf16t* __restrict__ vs, const bf16t* __restrict__ ka,
    const bf16t* __restrict__ va, const bf16t* __restrict__ kt,
    const bf16t* __restrict__ vt, const float* __restrict__ gate,
    bf16t* __restrict__ out) {
  __shared__ float sc[8][kNKEY + 7];
  __shared__ float qs[8][kHD];
  const int lane = threadIdx.x & 31;
  const int wave = threadIdx.x >> 5;
  const int gw = blockIdx.x * 8 + wave;       // 0..4095
  const int h  = gw % kNH;
  const int bt = gw / kNH;
  const int b  = bt / kT;
  const int t  = bt % kT;
  const float scale = 0.09449111825f;         // 1/sqrt(112)
  const float ratio = tanhf(gate[0]);

  const bf16t* qrow = q + ((size_t)(b * kT + t)) * kDIM + h * kHD;
  for (int d = lane; d < kHD; d += 32) qs[wave][d] = (float)qrow[d];
  __syncthreads();

  // scores
  for (int k = lane; k < kNKEY; k += 32) {
    const bf16t* krow; float mult = scale;
    if (k < kT)            krow = ks + ((size_t)(b * kT  + k)) * kDIM + h * kHD;
    else if (k < kT + kKA) krow = ka + ((size_t)(b * kKA + (k - kT))) * kDIM + h * kHD;
    else { krow = kt + ((size_t)(b * kKT + (k - kT - kKA))) * kDIM + h * kHD;
           mult = scale * ratio; }
    float s = 0.f;
    for (int d = 0; d < kHD; ++d) s += qs[wave][d] * (float)krow[d];
    sc[wave][k] = s * mult;
  }
  float m = -3.4e38f;
  for (int k = lane; k < kNKEY; k += 32) m = fmaxf(m, sc[wave][k]);
  for (int o = 16; o; o >>= 1) m = fmaxf(m, __shfl_xor(m, o, 32));
  float sum = 0.f;
  for (int k = lane; k < kNKEY; k += 32) {
    float e = expf(sc[wave][k] - m);
    sc[wave][k] = e;
    sum += e;
  }
  for (int o = 16; o; o >>= 1) sum += __shfl_xor(sum, o, 32);
  const float inv = 1.f / sum;
  __syncthreads();

  // weighted sum over V; lane owns dims {lane, lane+32, lane+64, lane+96<112}
  float o0 = 0.f, o1 = 0.f, o2 = 0.f, o3 = 0.f;
  for (int k = 0; k < kNKEY; ++k) {
    const float w = sc[wave][k] * inv;
    const bf16t* vrow;
    if (k < kT)            vrow = vs + ((size_t)(b * kT  + k)) * kDIM + h * kHD;
    else if (k < kT + kKA) vrow = va + ((size_t)(b * kKA + (k - kT))) * kDIM + h * kHD;
    else                   vrow = vt + ((size_t)(b * kKT + (k - kT - kKA))) * kDIM + h * kHD;
    o0 += w * (float)vrow[lane];
    o1 += w * (float)vrow[lane + 32];
    o2 += w * (float)vrow[lane + 64];
    if (lane < 16) o3 += w * (float)vrow[lane + 96];
  }
  bf16t* orow = out + ((size_t)(b * kT + t)) * kDIM + h * kHD;
  orow[lane]      = (bf16t)o0;
  orow[lane + 32] = (bf16t)o1;
  orow[lane + 64] = (bf16t)o2;
  if (lane < 16) orow[lane + 96] = (bf16t)o3;
}

// ---------------- residual + LayerNorm -> bf16 ------------------------------
__global__ void __launch_bounds__(256) resid_ln_kernel(
    const float* __restrict__ o, const float* __restrict__ x,
    const float* __restrict__ g, const float* __restrict__ bta,
    bf16t* __restrict__ yn) {
  const int row = blockIdx.x;
  const int tid = threadIdx.x;
  float s = 0.f, ss = 0.f;
  for (int i = tid; i < kDIM; i += 256) {
    const float y = o[(size_t)row * kDIM + i] + x[(size_t)row * kDIM + i];
    s += y; ss += y * y;
  }
  for (int off = 16; off; off >>= 1) {
    s  += __shfl_xor(s,  off, 32);
    ss += __shfl_xor(ss, off, 32);
  }
  __shared__ float sb[8], sb2[8];
  if ((tid & 31) == 0) { sb[tid >> 5] = s; sb2[tid >> 5] = ss; }
  __syncthreads();
  s = 0.f; ss = 0.f;
#pragma unroll
  for (int w = 0; w < 8; ++w) { s += sb[w]; ss += sb2[w]; }
  const float mu  = s / kDIM;
  const float var = ss / kDIM - mu * mu;
  const float inv = rsqrtf(var + 1e-5f);
  for (int i = tid; i < kDIM; i += 256) {
    const float y = o[(size_t)row * kDIM + i] + x[(size_t)row * kDIM + i];
    yn[(size_t)row * kDIM + i] = (bf16t)(((y - mu) * inv) * g[i] + bta[i]);
  }
}

// ---------------------------------------------------------------------------
extern "C" void kernel_launch(void* const* d_in, const int* in_sizes, int n_in,
                              void* d_out, int out_size, void* d_ws, size_t ws_size,
                              hipStream_t stream) {
  const float* x   = (const float*)d_in[0];
  const float* h_a = (const float*)d_in[1];
  const float* h_t = (const float*)d_in[2];
  const float* p   = (const float*)d_in[3];
  const float* W[9]; const float* bv[9];
  for (int i = 0; i < 9; ++i) {
    W[i]  = (const float*)d_in[4 + 2 * i];   // Wq,Wks,Wvs,Wka,Wva,Wkt,Wvt,Wo,Wf
    bv[i] = (const float*)d_in[5 + 2 * i];
  }
  const float* gate = (const float*)d_in[22];
  const float* ln_g = (const float*)d_in[23];
  const float* ln_b = (const float*)d_in[24];
  float* outp = (float*)d_out;

  const int MX = kB * kT;     // 512
  const int MA = kB * kKA;    // 4160
  const int MT = kB * kKT;    // 32768

  char* ws = (char*)d_ws;
  size_t off = 0;
  auto alloc = [&](size_t bytes) -> char* {
    char* pp = ws + off;
    off = (off + bytes + 255) & ~(size_t)255;
    return pp;
  };
  bf16t* Wt[9];
  for (int i = 0; i < 9; ++i) Wt[i] = (bf16t*)alloc((size_t)kDIM * kDIM * 2);
  bf16t* xbf    = (bf16t*)alloc((size_t)MX * kDIM * 2);
  bf16t* habf   = (bf16t*)alloc((size_t)MA * kDIM * 2);
  bf16t* htbf   = (bf16t*)alloc((size_t)MT * kDIM * 2);
  bf16t* qbf    = (bf16t*)alloc((size_t)MX * kDIM * 2);
  bf16t* ksbf   = (bf16t*)alloc((size_t)MX * kDIM * 2);
  bf16t* vsbf   = (bf16t*)alloc((size_t)MX * kDIM * 2);
  bf16t* kabf   = (bf16t*)alloc((size_t)MA * kDIM * 2);
  bf16t* vabf   = (bf16t*)alloc((size_t)MA * kDIM * 2);
  bf16t* ktbf   = (bf16t*)alloc((size_t)MT * kDIM * 2);
  bf16t* vtbf   = (bf16t*)alloc((size_t)MT * kDIM * 2);
  bf16t* attnbf = (bf16t*)alloc((size_t)MX * kDIM * 2);
  float* of32   = (float*)alloc((size_t)MX * kDIM * 4);
  bf16t* ynbf   = (bf16t*)alloc((size_t)MX * kDIM * 2);

  // 1) weights -> bf16 W^T
  dim3 tg(kDIM / 32, kDIM / 32);
  for (int i = 0; i < 9; ++i) wtrans_kernel<<<tg, 256, 0, stream>>>(W[i], Wt[i]);

  // 2) activations -> bf16 (+ h_ad concat)
  f2bf_kernel<<<1024, 256, 0, stream>>>(x,   xbf,  (long)MX * kDIM);
  f2bf_kernel<<<4096, 256, 0, stream>>>(h_t, htbf, (long)MT * kDIM);
  had_build_kernel<<<2048, 256, 0, stream>>>(h_a, p, habf);

  // 3) projections (WMMA)
  auto gm = [](int M) { return dim3((unsigned)((M + 127) / 128), kDIM / 128); };
  gemm_wmma_bf16<0, bf16t><<<gm(MX), 256, 0, stream>>>(xbf,  Wt[0], bv[0], qbf,  MX);
  gemm_wmma_bf16<0, bf16t><<<gm(MX), 256, 0, stream>>>(xbf,  Wt[1], bv[1], ksbf, MX);
  gemm_wmma_bf16<0, bf16t><<<gm(MX), 256, 0, stream>>>(xbf,  Wt[2], bv[2], vsbf, MX);
  gemm_wmma_bf16<0, bf16t><<<gm(MA), 256, 0, stream>>>(habf, Wt[3], bv[3], kabf, MA);
  gemm_wmma_bf16<0, bf16t><<<gm(MA), 256, 0, stream>>>(habf, Wt[4], bv[4], vabf, MA);
  gemm_wmma_bf16<0, bf16t><<<gm(MT), 256, 0, stream>>>(htbf, Wt[5], bv[5], ktbf, MT);
  gemm_wmma_bf16<0, bf16t><<<gm(MT), 256, 0, stream>>>(htbf, Wt[6], bv[6], vtbf, MT);

  // 4) RoPE (q, k_s with L=T; k_a with L=65; k_t with L=512)
  rope_kernel<<<((long)MX * 448 + 255) / 256, 256, 0, stream>>>(qbf,  MX, kT);
  rope_kernel<<<((long)MX * 448 + 255) / 256, 256, 0, stream>>>(ksbf, MX, kT);
  rope_kernel<<<((long)MA * 448 + 255) / 256, 256, 0, stream>>>(kabf, MA, kKA);
  rope_kernel<<<((long)MT * 448 + 255) / 256, 256, 0, stream>>>(ktbf, MT, kKT);

  // 5) attention: 4096 waves = 512 blocks of 8 waves
  attn_kernel<<<MX, 256, 0, stream>>>(qbf, ksbf, vsbf, kabf, vabf, ktbf, vtbf,
                                      gate, attnbf);

  // 6) Wo projection (f32 out), residual + LN, then Wf + ReLU -> d_out
  gemm_wmma_bf16<0, float><<<gm(MX), 256, 0, stream>>>(attnbf, Wt[7], bv[7], of32, MX);
  resid_ln_kernel<<<MX, 256, 0, stream>>>(of32, x, ln_g, ln_b, ynbf);
  gemm_wmma_bf16<1, float><<<gm(MX), 256, 0, stream>>>(ynbf, Wt[8], bv[8], outp, MX);
}
